// TestModel_10316511445289
// MI455X (gfx1250) — matrix operand order repaired and checked by reading, not compiled, not run on hardware
//
#include <hip/hip_runtime.h>
#include <hip/hip_bf16.h>

#define DM   768
#define TWOD 1536
#define MATSZ (TWOD * DM)

typedef __attribute__((ext_vector_type(16))) __bf16 v16bf;
typedef __attribute__((ext_vector_type(8)))  float  v8f;

union BF16x16 {
    v16bf v;
    __bf16 h[16];
    unsigned int u[8];
};

__device__ __forceinline__ float sigmoidf_(float x)
{
    return __fdividef(1.0f, 1.0f + __expf(-x));
}

// Block reduction: wave-level shfl_xor tree, then one LDS combine (2 barriers).
__device__ __forceinline__ float block_reduce_sum_256(float v)
{
    __shared__ float red[8];
    #pragma unroll
    for (int m = 16; m > 0; m >>= 1) v += __shfl_xor(v, m, 32);
    const int lane = threadIdx.x & 31, wv = threadIdx.x >> 5;
    if (lane == 0) red[wv] = v;
    __syncthreads();
    float r = 0.f;
    #pragma unroll
    for (int i = 0; i < 8; ++i) r += red[i];
    __syncthreads();
    return r;
}

// ---------------------------------------------------------------------------
// One-time weight prep: (nmat, 1536, 768) f32 row-major ->
//                       (nmat,  768, 1536) bf16, k-contiguous (transposed).
// ---------------------------------------------------------------------------
__global__ __launch_bounds__(256)
void wprep_kernel(const float* __restrict__ in, __bf16* __restrict__ outp)
{
    __shared__ float t[32][33];
    const long m = blockIdx.z;
    const int n0 = blockIdx.x * 32;       // over 768
    const int k0 = blockIdx.y * 32;       // over 1536
    const int tx = threadIdx.x & 31;
    const int ty = threadIdx.x >> 5;      // 0..7
    const float* src = in + m * MATSZ;
    __bf16* dst = outp + m * MATSZ;
    #pragma unroll
    for (int j = 0; j < 4; ++j)
        t[ty * 4 + j][tx] = src[(long)(k0 + ty * 4 + j) * DM + n0 + tx];
    __syncthreads();
    #pragma unroll
    for (int j = 0; j < 4; ++j)
        dst[(long)(n0 + ty * 4 + j) * TWOD + k0 + tx] = (__bf16)t[tx][ty * 4 + j];
}

// bf16 mirror of an f32 buffer
__global__ void f32_to_bf16_kernel(const float* __restrict__ in,
                                   __bf16* __restrict__ outp, int n)
{
    const int g = blockIdx.x * 256 + threadIdx.x;
    if (g < n) outp[g] = (__bf16)in[g];
}

// ---------------------------------------------------------------------------
// Dual GEMM: blockIdx.z selects {sel (sigmoid), proj (linear)}.
// out = act(A @ W + bias). A (bf16, M x 1536) assembled from two halves:
//   k in [0,768)    -> A0[b*bsA0 + r*rsA0 + k]
//   k in [768,1536) -> A1[b*bsA1 + r*rsA1 + (k-768)]
// WT bf16 (768 x 1536) k-contiguous (pre-transposed). out (M x 768) f32.
// Block: 128 threads = 4 waves, tile 128 rows x 128 cols.
// Wave: 32 rows x 128 cols -> 2 A frags x 8 B frags = 16 WMMA per k-step.
// No LDS, no barriers, no conversions in the hot loop.
// ---------------------------------------------------------------------------
__global__ __launch_bounds__(128)
void gemm_dual_kernel(const __bf16* __restrict__ A0, long rsA0, long bsA0,
                      const __bf16* __restrict__ A1, long rsA1, long bsA1,
                      const __bf16* __restrict__ Wsel, const float* __restrict__ bsel,
                      float* __restrict__ osel,
                      const __bf16* __restrict__ Wproj, const float* __restrict__ bproj,
                      float* __restrict__ oproj,
                      int M, int rowsPerBatch)
{
    const int which = blockIdx.z;                 // 0 = sel (sigmoid), 1 = proj
    const __bf16* WT   = which ? Wproj : Wsel;
    const float*  bias = which ? bproj : bsel;
    float*        out  = which ? oproj : osel;

    const int tid  = threadIdx.x;
    const int lane = tid & 31;
    const int wave = tid >> 5;
    const int n0   = blockIdx.y * 128;
    const int row0 = blockIdx.x * 128 + wave * 32;

    // A-fragment row pointers (both 16-lane halves cover m = lane & 15)
    const __bf16* pa0[2];
    const __bf16* pa1[2];
    #pragma unroll
    for (int h = 0; h < 2; ++h) {
        int arow = row0 + h * 16 + (lane & 15);
        if (arow >= M) arow = M - 1;             // clamp (stores are guarded)
        const int ab = arow / rowsPerBatch;
        const int ar = arow - ab * rowsPerBatch;
        pa0[h] = A0 + (long)ab * bsA0 + (long)ar * rsA0;
        pa1[h] = A1 + (long)ab * bsA1 + (long)ar * rsA1;
    }
    const int koff = (lane < 16) ? 0 : 8;        // A layout: 16-bit A 16x32
    const int kh   = (lane < 16) ? 0 : 8;        // B: same k order as A (8h+s, 16+8h+s)

    // B-fragment column pointers: lanes L and L+16 read alternating 16B chunks of
    // the same 64B k-run of column n = n0 + 16*i + (lane&15).
    const __bf16* wcol[8];
    #pragma unroll
    for (int i = 0; i < 8; ++i)
        wcol[i] = WT + (long)(n0 + i * 16 + (lane & 15)) * TWOD + kh;

    v8f acc[2][8] = {};

    for (int kt = 0; kt < TWOD; kt += 32) {
        // A fragments: 2x global_load_b128 each, already bf16
        BF16x16 afr[2];
        #pragma unroll
        for (int h = 0; h < 2; ++h) {
            const __bf16* ps = (kt < DM) ? (pa0[h] + kt) : (pa1[h] + (kt - DM));
            ((uint4*)afr[h].u)[0] = ((const uint4*)(ps + koff))[0];
            ((uint4*)afr[h].u)[1] = ((const uint4*)(ps + koff + 16))[0];
        }
        // Prefetch next k-run of this lane's first weight column
        if (kt + 32 < TWOD)
            __builtin_prefetch(wcol[0] + kt + 32, 0, 3);

        // 8 column tiles: B direct from pre-transposed bf16 weights (2x b128)
        #pragma unroll
        for (int i = 0; i < 8; ++i) {
            BF16x16 b;
            const uint4* s = (const uint4*)(wcol[i] + kt);
            ((uint4*)b.u)[0] = s[0];
            ((uint4*)b.u)[1] = s[2];
            acc[0][i] = __builtin_amdgcn_wmma_f32_16x16x32_bf16(
                false, afr[0].v, false, b.v, (short)0, acc[0][i], false, false);
            acc[1][i] = __builtin_amdgcn_wmma_f32_16x16x32_bf16(
                false, afr[1].v, false, b.v, (short)0, acc[1][i], false, false);
        }
    }

    // Epilogue: C layout — VGPR r: lanes 0-15 -> M=r, lanes 16-31 -> M=r+8, N=lane&15
    const int cn = lane & 15;
    #pragma unroll
    for (int h = 0; h < 2; ++h) {
        const int rbase = row0 + h * 16 + ((lane < 16) ? 0 : 8);
        #pragma unroll
        for (int i = 0; i < 8; ++i) {
            const int col = n0 + i * 16 + cn;
            const float bv = bias[col];
            #pragma unroll
            for (int r = 0; r < 8; ++r) {
                const int row = rbase + r;
                if (row < M) {
                    float vv = acc[h][i][r] + bv;
                    if (which == 0) vv = sigmoidf_(vv);
                    out[(long)row * DM + col] = vv;
                }
            }
        }
    }
}

// ---------------------------------------------------------------------------
// RMSNorm per row of 768: out = x * rsqrt(mean(x^2) + eps) * w
// ---------------------------------------------------------------------------
__global__ __launch_bounds__(256)
void rmsnorm_kernel(const float* __restrict__ xp, const float* __restrict__ w,
                    float* __restrict__ out)
{
    const long row = blockIdx.x;
    const int tid = threadIdx.x;
    const float* xr = xp + row * DM;
    float s = 0.f;
    #pragma unroll
    for (int i = 0; i < 3; ++i) { const float v = xr[tid + i * 256]; s += v * v; }
    const float tot = block_reduce_sum_256(s);
    const float scale = rsqrtf(tot * (1.0f / 768.0f) + 1.1920928955078125e-07f);
    float* orow = out + row * DM;
    #pragma unroll
    for (int i = 0; i < 3; ++i) { const int d = tid + i * 256; orow[d] = xr[d] * scale * w[d]; }
}

// ---------------------------------------------------------------------------
// Fused LIF conv + spike + gated fuse. One block per output row.
// conv: v_d = sum_{k<min(K,i+1)} tau_d^k * nrm[row-k][d]; spike = v>=th;
// denom = max(row spike count, 1).
// ---------------------------------------------------------------------------
__device__ __forceinline__ void lif_spikes_(const float* __restrict__ nrm,
                                            const float* __restrict__ tau,
                                            const float* __restrict__ th,
                                            long row, int kmax, int tid,
                                            float sp[3])
{
    float a0 = 0.f, a1 = 0.f, a2 = 0.f;
    float w0 = 1.f, w1 = 1.f, w2 = 1.f;
    const int d0 = tid, d1 = tid + 256, d2 = tid + 512;
    const float t0 = fminf(fmaxf(tau[d0], 1e-3f), 0.999f);
    const float t1 = fminf(fmaxf(tau[d1], 1e-3f), 0.999f);
    const float t2 = fminf(fmaxf(tau[d2], 1e-3f), 0.999f);
    const float* p = nrm + row * DM;
    for (int k = 0; k < kmax; ++k) {
        const float* pr = p - (long)k * DM;
        a0 += w0 * pr[d0]; w0 *= t0;
        a1 += w1 * pr[d1]; w1 *= t1;
        a2 += w2 * pr[d2]; w2 *= t2;
    }
    sp[0] = (a0 >= th[d0]) ? 1.0f : 0.0f;
    sp[1] = (a1 >= th[d1]) ? 1.0f : 0.0f;
    sp[2] = (a2 >= th[d2]) ? 1.0f : 0.0f;
}

__global__ __launch_bounds__(256)
void up_convfuse_kernel(const float* __restrict__ nrm, const float* __restrict__ tau,
                        const float* __restrict__ th, const float* __restrict__ gate,
                        const float* __restrict__ xp, const float* __restrict__ curr,
                        float* __restrict__ next, __bf16* __restrict__ next_bf,
                        int tpb, int K)
{
    const long row = blockIdx.x;
    const int tid = threadIdx.x;
    const int i = (int)(row % tpb);
    const int kmax = (K < i + 1) ? K : (i + 1);
    float sp[3];
    lif_spikes_(nrm, tau, th, row, kmax, tid, sp);
    const float inv = 1.0f / fmaxf(block_reduce_sum_256(sp[0] + sp[1] + sp[2]), 1.0f);
    #pragma unroll
    for (int c = 0; c < 3; ++c) {
        const int d = tid + c * 256;
        const long o = row * DM + d;
        const float res = 0.5f * (curr[2 * row * DM + d] + curr[(2 * row + 1) * DM + d]);
        const float outv = xp[o] + sp[c] * inv;
        const float g = gate[o];
        const float val = g * outv + (1.f - g) * res;
        next[o] = val;
        next_bf[o] = (__bf16)val;
    }
}

__global__ __launch_bounds__(256)
void down_convfuse_kernel(const float* __restrict__ nrm, const float* __restrict__ tau,
                          const float* __restrict__ th, const float* __restrict__ gate,
                          const float* __restrict__ xp, const float* __restrict__ ctx,
                          float* __restrict__ ctx_next, __bf16* __restrict__ ctx_next_bf,
                          int P, int K)
{
    const long row = blockIdx.x;
    const int tid = threadIdx.x;
    const int i = (int)(row % P);
    const int kmax = (K < i + 1) ? K : (i + 1);
    float sp[3];
    lif_spikes_(nrm, tau, th, row, kmax, tid, sp);
    const float inv = 1.0f / fmaxf(block_reduce_sum_256(sp[0] + sp[1] + sp[2]), 1.0f);
    const long b = row / P, j = row - b * P;
    const long e0 = (b * 2 * P + 2 * j) * DM;
    #pragma unroll
    for (int c = 0; c < 3; ++c) {
        const int d = tid + c * 256;
        const long o = row * DM + d;
        const float cv = ctx[o];
        const float outv = xp[o] + sp[c] * inv;
        const float g = gate[o];
        const float cr = g * outv + (1.f - g) * cv;
        ctx_next[e0 + d]         = cv;                 // even child keeps ctx
        ctx_next[e0 + DM + d]    = cr;                 // odd child gets ctx_r
        ctx_next_bf[e0 + d]      = (__bf16)cv;
        ctx_next_bf[e0 + DM + d] = (__bf16)cr;
    }
}

__global__ __launch_bounds__(256)
void leaf_convfuse_kernel(const float* __restrict__ nrm, const float* __restrict__ tau,
                          const float* __restrict__ th, const float* __restrict__ gate,
                          const float* __restrict__ xp, const float* __restrict__ x,
                          float* __restrict__ outp, int tpb, int K)
{
    const long row = blockIdx.x;
    const int tid = threadIdx.x;
    const int i = (int)(row % tpb);
    const int kmax = (K < i + 1) ? K : (i + 1);
    float sp[3];
    lif_spikes_(nrm, tau, th, row, kmax, tid, sp);
    const float inv = 1.0f / fmaxf(block_reduce_sum_256(sp[0] + sp[1] + sp[2]), 1.0f);
    #pragma unroll
    for (int c = 0; c < 3; ++c) {
        const int d = tid + c * 256;
        const long o = row * DM + d;
        const float g = gate[o];
        outp[o] = g * (xp[o] + sp[c] * inv) + (1.f - g) * x[o];
    }
}

__global__ void zero_kernel(float* __restrict__ p, int n)
{
    const int g = blockIdx.x * 256 + threadIdx.x;
    if (g < n) p[g] = 0.f;
}

// ---------------------------------------------------------------------------
extern "C" void kernel_launch(void* const* d_in, const int* in_sizes, int n_in,
                              void* d_out, int out_size, void* d_ws, size_t ws_size,
                              hipStream_t stream)
{
    (void)in_sizes; (void)n_in; (void)out_size; (void)ws_size;
    const float* x         = (const float*)d_in[0];
    const float* up_proj_W = (const float*)d_in[1];
    const float* up_proj_b = (const float*)d_in[2];
    const float* up_sel_W  = (const float*)d_in[3];
    const float* up_sel_b  = (const float*)d_in[4];
    const float* up_norm_w = (const float*)d_in[5];
    const float* up_th     = (const float*)d_in[6];
    const float* up_tau    = (const float*)d_in[7];
    const float* dn_proj_W = (const float*)d_in[8];
    const float* dn_proj_b = (const float*)d_in[9];
    const float* dn_sel_W  = (const float*)d_in[10];
    const float* dn_sel_b  = (const float*)d_in[11];
    const float* dn_norm_w = (const float*)d_in[12];
    const float* dn_th     = (const float*)d_in[13];
    const float* dn_tau    = (const float*)d_in[14];
    const float* lf_sel_W  = (const float*)d_in[15];
    const float* lf_sel_b  = (const float*)d_in[16];
    const float* lf_proj_W = (const float*)d_in[17];
    const float* lf_proj_b = (const float*)d_in[18];
    const float* lf_norm_w = (const float*)d_in[19];
    const float* lf_th     = (const float*)d_in[20];
    const float* lf_tau    = (const float*)d_in[21];

    float* ws = (float*)d_ws;
    long offRows[13]; offRows[1] = 0;
    for (int s = 1; s <= 11; ++s) offRows[s + 1] = offRows[s] + 4L * (2048 >> s);
    const long sumRows = offRows[12];             // 8188

    // f32 buffers
    float* sum  = ws;                             // summaries (levels 1..11)
    float* gate = sum  + sumRows * DM;
    float* xp   = gate + 8192L * DM;
    float* nrm  = xp   + 8192L * DM;
    float* ctxA = nrm  + 8192L * DM;
    float* ctxB = ctxA + 8192L * DM;
    // bf16 mirrors + weights
    __bf16* xbf    = (__bf16*)(ctxB + 8192L * DM);
    __bf16* sumbf  = xbf    + 8192L * DM;
    __bf16* ctxAbf = sumbf  + sumRows * DM;
    __bf16* ctxBbf = ctxAbf + 8192L * DM;
    __bf16* wbf    = ctxBbf + 8192L * DM;
    __bf16* w_up_proj = wbf;
    __bf16* w_up_sel  = wbf + 11L * MATSZ;
    __bf16* w_dn_proj = wbf + 22L * MATSZ;
    __bf16* w_dn_sel  = wbf + 33L * MATSZ;
    __bf16* w_lf_sel  = wbf + 44L * MATSZ;
    __bf16* w_lf_proj = wbf + 45L * MATSZ;

    // One-time per launch: transpose+convert all weights to bf16 [n][k]
    {
        dim3 b(256);
        wprep_kernel<<<dim3(24, 48, 11), b, 0, stream>>>(up_proj_W, w_up_proj);
        wprep_kernel<<<dim3(24, 48, 11), b, 0, stream>>>(up_sel_W,  w_up_sel);
        wprep_kernel<<<dim3(24, 48, 11), b, 0, stream>>>(dn_proj_W, w_dn_proj);
        wprep_kernel<<<dim3(24, 48, 11), b, 0, stream>>>(dn_sel_W,  w_dn_sel);
        wprep_kernel<<<dim3(24, 48, 1),  b, 0, stream>>>(lf_sel_W,  w_lf_sel);
        wprep_kernel<<<dim3(24, 48, 1),  b, 0, stream>>>(lf_proj_W, w_lf_proj);
    }
    // bf16 mirror of x; ctx starts as (B,1,D) zeros (f32 + bf16)
    f32_to_bf16_kernel<<<(8192 * DM + 255) / 256, 256, 0, stream>>>(x, xbf, 8192 * DM);
    zero_kernel<<<(4 * DM + 255) / 256, 256, 0, stream>>>(ctxA, 4 * DM);
    zero_kernel<<<(4 * DM / 2 + 255) / 256, 256, 0, stream>>>((float*)ctxAbf, 4 * DM / 2);

    // ---- up pass ----
    for (int l = 0; l < 11; ++l) {
        const int t = 2048 >> l, tpb = t >> 1, M = 4 * tpb;
        const float*  curr   = (l == 0) ? x   : (sum   + offRows[l] * DM);
        const __bf16* currbf = (l == 0) ? xbf : (sumbf + offRows[l] * DM);
        float*  next   = sum   + offRows[l + 1] * DM;
        __bf16* nextbf = sumbf + offRows[l + 1] * DM;
        const long rs = TWOD, bs = (long)t * DM;
        dim3 gg((M + 127) / 128, 6, 2);
        gemm_dual_kernel<<<gg, 128, 0, stream>>>(currbf, rs, bs, currbf + DM, rs, bs,
            w_up_sel + (long)l * MATSZ, up_sel_b + l * DM, gate,
            w_up_proj + (long)l * MATSZ, up_proj_b + l * DM, xp, M, tpb);
        rmsnorm_kernel<<<M, 256, 0, stream>>>(xp, up_norm_w + l * DM, nrm);
        const int K = tpb < 64 ? tpb : 64;
        up_convfuse_kernel<<<M, 256, 0, stream>>>(nrm, up_tau + l * DM, up_th + l * DM,
            gate, xp, curr, next, nextbf, tpb, K);
    }

    // ---- down pass ----
    float* ctx = ctxA; float* ctxN = ctxB;
    __bf16* cbf = ctxAbf; __bf16* cbfN = ctxBbf;
    for (int l = 11; l >= 1; --l) {
        const int P = 1 << (11 - l), M = 4 * P, i = l - 1;
        const __bf16* childbf = (i == 0) ? xbf : (sumbf + offRows[i] * DM);
        dim3 gg((M + 127) / 128, 6, 2);
        gemm_dual_kernel<<<gg, 128, 0, stream>>>(cbf, DM, (long)P * DM,
            childbf, TWOD, (long)2 * P * DM,
            w_dn_sel + (long)i * MATSZ, dn_sel_b + i * DM, gate,
            w_dn_proj + (long)i * MATSZ, dn_proj_b + i * DM, xp, M, P);
        rmsnorm_kernel<<<M, 256, 0, stream>>>(xp, dn_norm_w + i * DM, nrm);
        const int K = P < 64 ? P : 64;
        down_convfuse_kernel<<<M, 256, 0, stream>>>(nrm, dn_tau + i * DM, dn_th + i * DM,
            gate, xp, ctx, ctxN, cbfN, P, K);
        float* tf = ctx; ctx = ctxN; ctxN = tf;
        __bf16* tb = cbf; cbf = cbfN; cbfN = tb;
    }

    // ---- leaf fusion ----
    {
        const int M = 8192, tpb = 2048;
        dim3 gg(M / 128, 6, 2);
        gemm_dual_kernel<<<gg, 128, 0, stream>>>(xbf, DM, (long)2048 * DM,
            cbf, DM, (long)2048 * DM,
            w_lf_sel, lf_sel_b, gate,
            w_lf_proj, lf_proj_b, xp, M, tpb);
        rmsnorm_kernel<<<M, 256, 0, stream>>>(xp, lf_norm_w, nrm);
        leaf_convfuse_kernel<<<M, 256, 0, stream>>>(nrm, lf_tau, lf_th,
            gate, xp, x, (float*)d_out, tpb, 64);
    }
}
